// GDTLayer_67654324847234
// MI455X (gfx1250) — matrix operand order repaired and checked by reading, not compiled, not run on hardware
//
#include <hip/hip_runtime.h>
#include <hip/hip_bf16.h>
#include <cstdint>

// ---- problem constants (match reference) ----
static constexpr int kN   = 50000;
static constexpr int kD   = 16;     // neighbors
static constexpr int kDIM = 128;
static constexpr int kH   = 8;
static constexpr int kHD  = 16;
static constexpr float kALPHA = 0.1f;
static constexpr int kTOPK = 5;
static constexpr float kSLOPE = 0.2f;
static constexpr float kLNEPS = 1e-5f;
// log(17)/16
static constexpr float kSCALE = 2.8332133440562162f / 16.0f;

static constexpr int kKC   = 128;        // K-chunk staged in LDS
static constexpr int kPAD  = 8;          // halves of per-column padding (bank skew)
static constexpr int kLDBL = kKC + kPAD; // LDS column stride in halves

typedef __attribute__((ext_vector_type(16))) _Float16 v16h;
typedef __attribute__((ext_vector_type(8)))  _Float16 v8h;
typedef __attribute__((ext_vector_type(8)))  float    v8f;

// ---------------------------------------------------------------------------
// CDNA5 async global->LDS copy (16B per lane) + wait   [ISA 15.18.3 / ch.10]
// ---------------------------------------------------------------------------
__device__ __forceinline__ void async_copy16(void* lds_ptr, const void* gptr) {
  uint32_t lds_off = (uint32_t)(uintptr_t)lds_ptr;     // LDS generic addr low 32 = offset
  uint64_t ga      = (uint64_t)(uintptr_t)gptr;
  asm volatile("global_load_async_to_lds_b128 %0, %1, off"
               :: "v"(lds_off), "v"(ga) : "memory");
}
__device__ __forceinline__ void wait_async0() {
  asm volatile("s_wait_asynccnt 0x0" ::: "memory");
}

// ---------------------------------------------------------------------------
// fragment helpers (CDNA5 WMMA f16 16x16x32 layouts, ISA 7.12.2, wave32)
// ---------------------------------------------------------------------------
__device__ __forceinline__ v16h make_v16h(v8h lo, v8h hi) {
  return __builtin_shufflevector(lo, hi, 0,1,2,3,4,5,6,7,8,9,10,11,12,13,14,15);
}

// A: 16x32 (MxK), row-major, leading dim lda.
__device__ __forceinline__ v16h load_a_frag(const _Float16* __restrict__ A, int lda,
                                            int m0, int k0, int lane) {
  int m  = m0 + (lane & 15);
  int kb = (lane >> 4) << 3;
  const _Float16* p = A + (size_t)m * lda + k0 + kb;
  v8h lo = *(const v8h*)(p);
  v8h hi = *(const v8h*)(p + 16);
  return make_v16h(lo, hi);
}

// B frag from LDS-staged panel: Blds[col][kLDBL], col = local col 0..63.
// lane<16: N=lane, K=0..7,16..23 ; lane>=16: N=lane-16, K=8..15,24..31 (matches A).
__device__ __forceinline__ v16h load_b_frag_lds(const _Float16* Blds, int ncol0,
                                                int k0, int lane) {
  const _Float16* p = Blds + (size_t)(ncol0 + (lane & 15)) * kLDBL
                           + k0 + ((lane >> 4) << 3);
  v8h lo = *(const v8h*)(p);
  v8h hi = *(const v8h*)(p + 16);
  return make_v16h(lo, hi);
}

// ---------------------------------------------------------------------------
// GEMM core: block = 8 waves = 8 consecutive M-tiles, one shared 64-col group.
// B panel (64 x K, from transposed weights BT[ncols][K]) staged in LDS in
// K-chunks of 128 via async loads; all waves consume it via ds_load_b128.
// ---------------------------------------------------------------------------
template <int K>
__device__ __forceinline__ void gemm_core_lds(const _Float16* __restrict__ A,
                                              const _Float16* __restrict__ BT,
                                              _Float16* Blds, int nG, int mT,
                                              bool valid, int lane, int tid,
                                              v8f acc[4]) {
  for (int kc = 0; kc < K; kc += kKC) {
    // stage 64 cols x 128 halves (16 KB): 1024 x 16B units, 256 threads x 4
#pragma unroll
    for (int i = 0; i < 4; ++i) {
      int u   = tid + 256 * i;       // 16B unit
      int col = u >> 4;              // 16 units (256B) per column
      int hof = (u & 15) << 3;       // half offset within column chunk
      const _Float16* g = BT + (size_t)(nG * 64 + col) * K + kc + hof;
      async_copy16(&Blds[(size_t)col * kLDBL + hof], g);
    }
    wait_async0();
    __syncthreads();
    if (valid) {
#pragma unroll
      for (int k0 = 0; k0 < kKC; k0 += 32) {
        v16h a = load_a_frag(A, K, mT * 16, kc + k0, lane);
#pragma unroll
        for (int t = 0; t < 4; ++t) {
          v16h b = load_b_frag_lds(Blds, t * 16, k0, lane);
          acc[t] = __builtin_amdgcn_wmma_f32_16x16x32_f16(
              false, a, false, b, (short)0, acc[t], false, false);
        }
      }
    }
    __syncthreads();   // protect LDS reuse for next K-chunk
  }
}

// ---------------------------------------------------------------------------
// 0) convert weights to f16, TRANSPOSED (BT[ncol][K])
// ---------------------------------------------------------------------------
__global__ void convw_kernel(const float* __restrict__ Wh, const float* __restrict__ Wt,
                             const float* __restrict__ We, const float* __restrict__ Wf1,
                             const float* __restrict__ Wf2,
                             _Float16* __restrict__ WprojT, _Float16* __restrict__ Wf1T,
                             _Float16* __restrict__ Wf2T) {
  int id = blockIdx.x * blockDim.x + threadIdx.x;
  const int n0 = 384 * 128, n1 = n0 + 512 * 128, n2 = n1 + 128 * 512;
  if (id < n0) {
    int j = id >> 7, k = id & 127;                 // j: output col 0..383
    const float* W = (j < 128) ? Wh : ((j < 256) ? Wt : We);
    WprojT[id] = (_Float16)W[(size_t)k * 128 + (j & 127)];
  } else if (id < n1) {
    int t = id - n0; int j = t >> 7, k = t & 127;  // Wf1: [128][512] -> T[512][128]
    Wf1T[t] = (_Float16)Wf1[(size_t)k * 512 + j];
  } else if (id < n2) {
    int t = id - n1; int j = t >> 9, k = t & 511;  // Wf2: [512][128] -> T[128][512]
    Wf2T[t] = (_Float16)Wf2[(size_t)k * 128 + j];
  }
}

// ---------------------------------------------------------------------------
// 1) LayerNorm: wave-per-row, 4 floats/lane, shfl reductions; f16 output
// ---------------------------------------------------------------------------
__global__ void ln1_kernel(const float* __restrict__ feat, const float* __restrict__ g,
                           const float* __restrict__ b, _Float16* __restrict__ xo) {
  int w = threadIdx.x >> 5, l = threadIdx.x & 31;
  int n = blockIdx.x * 8 + w;
  if (n >= kN) return;
  const float* row = feat + (size_t)n * kDIM;
  float v[4];
  float s = 0.f;
#pragma unroll
  for (int i = 0; i < 4; ++i) { v[i] = row[l * 4 + i]; s += v[i]; }
#pragma unroll
  for (int off = 16; off > 0; off >>= 1) s += __shfl_xor(s, off, 32);
  float mu = s * (1.0f / kDIM);
  float q = 0.f;
#pragma unroll
  for (int i = 0; i < 4; ++i) { float d = v[i] - mu; q += d * d; }
#pragma unroll
  for (int off = 16; off > 0; off >>= 1) q += __shfl_xor(q, off, 32);
  float rinv = rsqrtf(q * (1.0f / kDIM) + kLNEPS);
  _Float16* out = xo + (size_t)n * kDIM;
#pragma unroll
  for (int i = 0; i < 4; ++i) {
    int c = l * 4 + i;
    out[c] = (_Float16)((v[i] - mu) * rinv * g[c] + b[c]);
  }
}

// 1b) rst = h_final + feat ; y = LN(rst)  (emit rst f32 AND y f16)
__global__ void ln2_kernel(const float* __restrict__ hfin, const float* __restrict__ feat,
                           const float* __restrict__ g, const float* __restrict__ b,
                           float* __restrict__ rst, _Float16* __restrict__ yo) {
  int w = threadIdx.x >> 5, l = threadIdx.x & 31;
  int n = blockIdx.x * 8 + w;
  if (n >= kN) return;
  const float* hr = hfin + (size_t)n * kDIM;
  const float* fr = feat + (size_t)n * kDIM;
  float v[4];
  float s = 0.f;
#pragma unroll
  for (int i = 0; i < 4; ++i) { int c = l * 4 + i; v[i] = hr[c] + fr[c]; s += v[i]; }
#pragma unroll
  for (int off = 16; off > 0; off >>= 1) s += __shfl_xor(s, off, 32);
  float mu = s * (1.0f / kDIM);
  float q = 0.f;
#pragma unroll
  for (int i = 0; i < 4; ++i) { float d = v[i] - mu; q += d * d; }
#pragma unroll
  for (int off = 16; off > 0; off >>= 1) q += __shfl_xor(q, off, 32);
  float rinv = rsqrtf(q * (1.0f / kDIM) + kLNEPS);
  float* rr = rst + (size_t)n * kDIM;
  _Float16* yy = yo + (size_t)n * kDIM;
#pragma unroll
  for (int i = 0; i < 4; ++i) {
    int c = l * 4 + i;
    rr[c] = v[i];
    yy[c] = (_Float16)((v[i] - mu) * rinv * g[c] + b[c]);
  }
}

// ---------------------------------------------------------------------------
// 2) projections: [fh|ft|fe] = x @ [Wh|Wt|We]  (K=128, 6 col-groups of 64)
// ---------------------------------------------------------------------------
__global__ void proj_gemm(const _Float16* __restrict__ x, const _Float16* __restrict__ WT,
                          float* __restrict__ fh, float* __restrict__ ft,
                          float* __restrict__ fe, int mBlocks) {
  __shared__ _Float16 Blds[64 * kLDBL];
  int nG = blockIdx.x / mBlocks;
  int mB = blockIdx.x % mBlocks;
  int wv = threadIdx.x >> 5, lane = threadIdx.x & 31;
  int mT = mB * 8 + wv;
  bool valid = mT < (kN / 16);
  v8f acc[4];
#pragma unroll
  for (int t = 0; t < 4; ++t) acc[t] = (v8f){0,0,0,0,0,0,0,0};
  gemm_core_lds<kDIM>(x, WT, Blds, nG, mT, valid, lane, threadIdx.x, acc);
  if (!valid) return;
  int rbase = mT * 16 + ((lane >> 4) << 3);
  float* bufs[3] = {fh, ft, fe};
#pragma unroll
  for (int t = 0; t < 4; ++t) {
    int cg = nG * 64 + t * 16 + (lane & 15);
    float* dst = bufs[cg >> 7];
    int cc = cg & 127;
#pragma unroll
    for (int r = 0; r < 8; ++r)
      dst[(size_t)(rbase + r) * kDIM + cc] = acc[t][r];
  }
}

// ---------------------------------------------------------------------------
// 3) scores + softmax(D) + top-5 threshold + renorm -> a_n [N][H][D]
// ---------------------------------------------------------------------------
__global__ void score_kernel(const float* __restrict__ fh, const float* __restrict__ ft,
                             const float* __restrict__ attn, const int* __restrict__ src,
                             float* __restrict__ an_out) {
  __shared__ float sc[8][kD * kH];  // [d*8+h]
  int w = threadIdx.x >> 5, l = threadIdx.x & 31;
  int n = blockIdx.x * 8 + w;
  if (n >= kN) return;
#pragma unroll
  for (int i = 0; i < 4; ++i) {
    int p = l + 32 * i;
    int d = p >> 3, h = p & 7;
    int sidx = src[(size_t)n * kD + d];
    const float* fhr = fh + (size_t)sidx * kDIM + h * kHD;
    const float* ftr = ft + (size_t)n * kDIM + h * kHD;
    const float* at  = attn + h * kHD;
    float acc = 0.f;
#pragma unroll
    for (int k = 0; k < kHD; ++k) {
      float e = fhr[k] + ftr[k];
      e = (e > 0.f) ? e : kSLOPE * e;
      acc += e * at[k];
    }
    sc[w][p] = acc * kSCALE;
  }
  __syncthreads();
  if (l < kH) {
    int h = l;
    float v[kD];
    float m = -3.0e38f;
#pragma unroll
    for (int d = 0; d < kD; ++d) { v[d] = sc[w][d * 8 + h]; m = fmaxf(m, v[d]); }
    float e[kD];
    float sum = 0.f;
#pragma unroll
    for (int d = 0; d < kD; ++d) { e[d] = __expf(v[d] - m); sum += e[d]; }
    float inv = 1.0f / sum;
#pragma unroll
    for (int d = 0; d < kD; ++d) e[d] *= inv;
    float tmp[kD];
#pragma unroll
    for (int d = 0; d < kD; ++d) tmp[d] = e[d];
    float thr = 0.f;
#pragma unroll
    for (int t = 0; t < kTOPK; ++t) {
      float mx = -1.f; int mi = 0;
#pragma unroll
      for (int d = 0; d < kD; ++d) if (tmp[d] > mx) { mx = tmp[d]; mi = d; }
      thr = mx; tmp[mi] = -2.f;
    }
    float tsum = 0.f;
#pragma unroll
    for (int d = 0; d < kD; ++d) if (e[d] >= thr) tsum += e[d];
    float itn = 1.0f / tsum;
    float* ao = an_out + (size_t)n * (kH * kD) + h * kD;
#pragma unroll
    for (int d = 0; d < kD; ++d) ao[d] = (e[d] >= thr) ? e[d] * itn : 0.f;
  }
}

// ---------------------------------------------------------------------------
// 4) one propagation hop: hout = (1-a)*agg + a*fe ; wave per node
// ---------------------------------------------------------------------------
__global__ void hop_kernel(const float* __restrict__ hin, const float* __restrict__ fe,
                           const float* __restrict__ an_g, const int* __restrict__ src,
                           float* __restrict__ hout) {
  __shared__ int   s_src[8][kD];
  __shared__ float s_an[8][kH * kD];
  int w = threadIdx.x >> 5, l = threadIdx.x & 31;
  int n = blockIdx.x * 8 + w;
  int nn = (n < kN) ? n : 0;
  if (l < kD) s_src[w][l] = src[(size_t)nn * kD + l];
#pragma unroll
  for (int i = 0; i < 4; ++i) s_an[w][l + 32 * i] = an_g[(size_t)nn * 128 + l + 32 * i];
  __syncthreads();
  if (n >= kN) return;
#pragma unroll
  for (int i = 0; i < 4; ++i) {
    int q = l + 32 * i;            // q = h*16 + f
    int h = q >> 4;
    float acc = 0.f;
#pragma unroll
    for (int d = 0; d < kD; ++d)
      acc += s_an[w][h * kD + d] * hin[(size_t)s_src[w][d] * kDIM + q];
    hout[(size_t)n * kDIM + q] = (1.0f - kALPHA) * acc + kALPHA * fe[(size_t)n * kDIM + q];
  }
}

// ---------------------------------------------------------------------------
// 5) FFN1: mid = relu(y @ Wf1 + b1)  (K=128, 8 col-groups), f16 out
// ---------------------------------------------------------------------------
__global__ void ffn1_gemm(const _Float16* __restrict__ y, const _Float16* __restrict__ W1T,
                          const float* __restrict__ b1, _Float16* __restrict__ mid,
                          int mBlocks) {
  __shared__ _Float16 Blds[64 * kLDBL];
  int nG = blockIdx.x / mBlocks;
  int mB = blockIdx.x % mBlocks;
  int wv = threadIdx.x >> 5, lane = threadIdx.x & 31;
  int mT = mB * 8 + wv;
  bool valid = mT < (kN / 16);
  v8f acc[4];
#pragma unroll
  for (int t = 0; t < 4; ++t) acc[t] = (v8f){0,0,0,0,0,0,0,0};
  gemm_core_lds<kDIM>(y, W1T, Blds, nG, mT, valid, lane, threadIdx.x, acc);
  if (!valid) return;
  int rbase = mT * 16 + ((lane >> 4) << 3);
#pragma unroll
  for (int t = 0; t < 4; ++t) {
    int cg = nG * 64 + t * 16 + (lane & 15);
    float bb = b1[cg];
#pragma unroll
    for (int r = 0; r < 8; ++r) {
      float v = acc[t][r] + bb;
      v = (v > 0.f) ? v : 0.f;
      mid[(size_t)(rbase + r) * 512 + cg] = (_Float16)v;
    }
  }
}

// 6) FFN2: out = mid @ Wf2 + b2 + rst  (K=512, 2 col-groups), f32 out
__global__ void ffn2_gemm(const _Float16* __restrict__ mid, const _Float16* __restrict__ W2T,
                          const float* __restrict__ b2, const float* __restrict__ rst,
                          float* __restrict__ out, int mBlocks) {
  __shared__ _Float16 Blds[64 * kLDBL];
  int nG = blockIdx.x / mBlocks;
  int mB = blockIdx.x % mBlocks;
  int wv = threadIdx.x >> 5, lane = threadIdx.x & 31;
  int mT = mB * 8 + wv;
  bool valid = mT < (kN / 16);
  v8f acc[4];
#pragma unroll
  for (int t = 0; t < 4; ++t) acc[t] = (v8f){0,0,0,0,0,0,0,0};
  gemm_core_lds<512>(mid, W2T, Blds, nG, mT, valid, lane, threadIdx.x, acc);
  if (!valid) return;
  int rbase = mT * 16 + ((lane >> 4) << 3);
#pragma unroll
  for (int t = 0; t < 4; ++t) {
    int cg = nG * 64 + t * 16 + (lane & 15);
    float bb = b2[cg];
#pragma unroll
    for (int r = 0; r < 8; ++r) {
      size_t idx = (size_t)(rbase + r) * kDIM + cg;
      out[idx] = acc[t][r] + bb + rst[idx];
    }
  }
}

// ---------------------------------------------------------------------------
extern "C" void kernel_launch(void* const* d_in, const int* in_sizes, int n_in,
                              void* d_out, int out_size, void* d_ws, size_t ws_size,
                              hipStream_t stream) {
  (void)in_sizes; (void)n_in; (void)out_size; (void)ws_size;
  const float* feat   = (const float*)d_in[0];
  const float* W_head = (const float*)d_in[1];
  const float* W_tail = (const float*)d_in[2];
  const float* W_ent  = (const float*)d_in[3];
  const float* attn   = (const float*)d_in[4];
  const float* g1     = (const float*)d_in[5];
  const float* b1     = (const float*)d_in[6];
  const float* g2     = (const float*)d_in[7];
  const float* b2     = (const float*)d_in[8];
  const float* W_ff1  = (const float*)d_in[9];
  const float* b_ff1  = (const float*)d_in[10];
  const float* W_ff2  = (const float*)d_in[11];
  const float* b_ff2  = (const float*)d_in[12];
  const int*   src    = (const int*)d_in[13];
  float* out = (float*)d_out;

  // workspace arena (with aliasing)
  char* w = (char*)d_ws;
  const size_t rowF = (size_t)kN * kDIM * sizeof(float);      // 25.6 MB
  float* fh  = (float*)w;                w += rowF;           // -> hA after scores
  float* ft  = (float*)w;                w += rowF;           // -> hB after scores
  float* fe  = (float*)w;                w += rowF;           // \ reused as `mid`
  float* an  = (float*)w;                w += rowF;           // /   (51.2 MB) for FFN
  float* rst = (float*)w;                w += rowF;
  _Float16* xh = (_Float16*)w;           w += (size_t)kN * kDIM * sizeof(_Float16); // x, later y
  _Float16* WprojT = (_Float16*)w;       w += (size_t)384 * 128 * sizeof(_Float16);
  _Float16* Wf1T   = (_Float16*)w;       w += (size_t)512 * 128 * sizeof(_Float16);
  _Float16* Wf2T   = (_Float16*)w;       w += (size_t)128 * 512 * sizeof(_Float16);
  float* hA = fh;
  float* hB = ft;
  _Float16* mid = (_Float16*)fe;   // 51.2 MB spans fe+an region
  _Float16* yh  = xh;

  const int nodeBlocks = (kN + 7) / 8;        // 6250, wave-per-node kernels
  const int mBlocks    = (kN / 16 + 7) / 8;   // 391 blocks of 8 M-tiles

  // 0) weights -> f16 transposed
  {
    int total = 384 * 128 + 512 * 128 + 128 * 512;
    convw_kernel<<<(total + 255) / 256, 256, 0, stream>>>(
        W_head, W_tail, W_ent, W_ff1, W_ff2, WprojT, Wf1T, Wf2T);
  }
  // 1) x = LN(feat)
  ln1_kernel<<<nodeBlocks, 256, 0, stream>>>(feat, g1, b1, xh);
  // 2) fh/ft/fe = x @ [Wh|Wt|We]
  proj_gemm<<<6 * mBlocks, 256, 0, stream>>>(xh, WprojT, fh, ft, fe, mBlocks);
  // 3) attention coefficients
  score_kernel<<<nodeBlocks, 256, 0, stream>>>(fh, ft, attn, src, an);
  // 4) 5 hops: fe -> hA -> hB -> hA -> hB -> hA
  hop_kernel<<<nodeBlocks, 256, 0, stream>>>(fe, fe, an, src, hA);
  hop_kernel<<<nodeBlocks, 256, 0, stream>>>(hA, fe, an, src, hB);
  hop_kernel<<<nodeBlocks, 256, 0, stream>>>(hB, fe, an, src, hA);
  hop_kernel<<<nodeBlocks, 256, 0, stream>>>(hA, fe, an, src, hB);
  hop_kernel<<<nodeBlocks, 256, 0, stream>>>(hB, fe, an, src, hA);
  // 5) rst = hA + feat ; y = LN(rst)
  ln2_kernel<<<nodeBlocks, 256, 0, stream>>>(hA, feat, g2, b2, rst, yh);
  // 6) FFN (mid overwrites fe/an, both dead now)
  ffn1_gemm<<<8 * mBlocks, 256, 0, stream>>>(yh, Wf1T, b_ff1, mid, mBlocks);
  ffn2_gemm<<<2 * mBlocks, 256, 0, stream>>>(mid, Wf2T, b_ff2, rst, out, mBlocks);
}
